// ScaleQuant_64261300682783
// MI455X (gfx1250) — 8-bit matrix operand order repaired and checked by reading, compiled, not run on hardware
//
#include <hip/hip_runtime.h>
#include <math.h>


typedef __attribute__((ext_vector_type(8))) int v8i;

#define OFF_ENC   0
#define OFF_SCAL  64
#define OFF_ROWW  256
#define OFF_ROWX  16640
#define OFF_Q1    49664
#define OFF_XQ    16827136

#define S_S1    0
#define S_Z1    1
#define S_S2    2
#define S_Z2    3
#define S_S1S2  4
#define S_Z1Z2K 5
#define S_INVS1 6
#define S_RCPS2 7
#define S_MNW   8

__device__ __forceinline__ unsigned encf(float f) {
    unsigned u = __float_as_uint(f);
    return (u & 0x80000000u) ? ~u : (u | 0x80000000u);
}
__device__ __forceinline__ float decf(unsigned u) {
    u = (u & 0x80000000u) ? (u & 0x7FFFFFFFu) : ~u;
    return __uint_as_float(u);
}

__global__ void init_scalars(unsigned* enc) {
    enc[0] = 0xFFFFFFFFu;
    enc[1] = 0u;
    enc[2] = 0xFFFFFFFFu;
    enc[3] = 0u;
}

__global__ __launch_bounds__(256) void minmax_kernel(const float* __restrict__ p,
                                                     long long n4,
                                                     unsigned* encMin,
                                                     unsigned* encMax) {
    __shared__ float smn[256], smx[256];
    const float4* p4 = (const float4*)p;
    float mn = 3.4e38f, mx = -3.4e38f;
    long long stride = (long long)gridDim.x * blockDim.x;
    for (long long i = (long long)blockIdx.x * blockDim.x + threadIdx.x; i < n4; i += stride) {
        float4 v = p4[i];
        mn = fminf(mn, fminf(fminf(v.x, v.y), fminf(v.z, v.w)));
        mx = fmaxf(mx, fmaxf(fmaxf(v.x, v.y), fmaxf(v.z, v.w)));
    }
    int t = threadIdx.x;
    smn[t] = mn; smx[t] = mx;
    __syncthreads();
    for (int s = 128; s > 0; s >>= 1) {
        if (t < s) {
            smn[t] = fminf(smn[t], smn[t + s]);
            smx[t] = fmaxf(smx[t], smx[t + s]);
        }
        __syncthreads();
    }
    if (t == 0) {
        atomicMin(encMin, encf(smn[0]));
        atomicMax(encMax, encf(smx[0]));
    }
}

__global__ void params_kernel(const unsigned* enc, float* scal, int K) {
    float mnW = decf(enc[0]), mxW = decf(enc[1]);
    float mnX = decf(enc[2]), mxX = decf(enc[3]);
    const float n = 255.0f;
    float S1 = (mxW - mnW) / n;
    float Z1 = rintf(n * mnW / (mnW - mxW));
    float S2 = (mxX - mnX) / n;
    float Z2 = rintf(n * mnX / (mnX - mxX));
    scal[S_S1] = S1;
    scal[S_Z1] = Z1;
    scal[S_S2] = S2;
    scal[S_Z2] = Z2;
    scal[S_S1S2] = S1 * S2;
    scal[S_Z1Z2K] = Z1 * Z2 * (float)K;
    scal[S_INVS1] = n / (mxW - mnW);
    scal[S_RCPS2] = 1.0f / S2;
    scal[S_MNW] = mnW;
}

__global__ __launch_bounds__(256) void quantW_kernel(const float* __restrict__ W,
                                                     unsigned char* __restrict__ q1,
                                                     int* __restrict__ rowW,
                                                     const float* __restrict__ scal,
                                                     int K) {
    __shared__ int ssum[256];
    const int row = blockIdx.x;
    const float mn = scal[S_MNW];
    const float inv = scal[S_INVS1];
    const float* w = W + (size_t)row * K;
    const int base = threadIdx.x * 16;
    int sum = 0;
    unsigned packed[4];
#pragma unroll
    for (int g = 0; g < 4; ++g) {
        float4 v = *(const float4*)(w + base + g * 4);
        int q0 = (int)rintf((v.x - mn) * inv);
        int q1v = (int)rintf((v.y - mn) * inv);
        int q2 = (int)rintf((v.z - mn) * inv);
        int q3 = (int)rintf((v.w - mn) * inv);
        sum += q0 + q1v + q2 + q3;
        packed[g] = (unsigned)(q0 & 255) | ((unsigned)(q1v & 255) << 8) |
                    ((unsigned)(q2 & 255) << 16) | ((unsigned)(q3 & 255) << 24);
    }
    *(uint4*)(q1 + (size_t)row * K + base) = make_uint4(packed[0], packed[1], packed[2], packed[3]);
    int t = threadIdx.x;
    ssum[t] = sum;
    __syncthreads();
    for (int s = 128; s > 0; s >>= 1) {
        if (t < s) ssum[t] += ssum[t + s];
        __syncthreads();
    }
    if (t == 0) rowW[row] = ssum[0];
}

__global__ __launch_bounds__(256) void quantX_kernel(const float* __restrict__ X,
                                                     unsigned char* __restrict__ xq,
                                                     int* __restrict__ rowX,
                                                     const float* __restrict__ scal,
                                                     int K) {
    __shared__ int ssum[256];
    const int row = blockIdx.x;
    const float rcpS2 = scal[S_RCPS2];
    const float Z2 = scal[S_Z2];
    const float* x = X + (size_t)row * K;
    const int base = threadIdx.x * 16;
    int sum = 0;
    unsigned packed[4];
#pragma unroll
    for (int g = 0; g < 4; ++g) {
        float4 v = *(const float4*)(x + base + g * 4);
        int q0 = (int)fminf(255.0f, fmaxf(0.0f, rintf(v.x * rcpS2 + Z2)));
        int q1v = (int)fminf(255.0f, fmaxf(0.0f, rintf(v.y * rcpS2 + Z2)));
        int q2 = (int)fminf(255.0f, fmaxf(0.0f, rintf(v.z * rcpS2 + Z2)));
        int q3 = (int)fminf(255.0f, fmaxf(0.0f, rintf(v.w * rcpS2 + Z2)));
        sum += q0 + q1v + q2 + q3;
        packed[g] = (unsigned)q0 | ((unsigned)q1v << 8) | ((unsigned)q2 << 16) | ((unsigned)q3 << 24);
    }
    *(uint4*)(xq + (size_t)row * K + base) = make_uint4(packed[0], packed[1], packed[2], packed[3]);
    int t = threadIdx.x;
    ssum[t] = sum;
    __syncthreads();
    for (int s = 128; s > 0; s >>= 1) {
        if (t < s) ssum[t] += ssum[t + s];
        __syncthreads();
    }
    if (t == 0) rowX[row] = ssum[0];
}

#define BM 128
#define BN 128
#define BK 64

__global__ __launch_bounds__(256) void gemm_iu8_kernel(
        const unsigned char* __restrict__ Aq,
        const unsigned char* __restrict__ Bq,
        const int* __restrict__ rowX,
        const int* __restrict__ rowW,
        const float* __restrict__ bias,
        const float* __restrict__ scal,
        float* __restrict__ out,
        int M, int N, int K) {
    __shared__ uint4 As[2][BM][BK / 16];
    __shared__ uint4 Bs[2][BN][BK / 16];

    const int tid = threadIdx.x;
    const int lane = tid & 31;
    const int wave = tid >> 5;
    const int laneLo = lane & 15;
    const int laneHi = lane >> 4;

    const int mBlock = blockIdx.y * BM;
    const int nBlock = blockIdx.x * BN;

    const int waveM = (wave >> 1) * 32;
    const int waveN = (wave & 1) * 64;

    v8i acc[2][4];
#pragma unroll
    for (int mi = 0; mi < 2; ++mi)
#pragma unroll
        for (int ni = 0; ni < 4; ++ni)
            acc[mi][ni] = (v8i){0, 0, 0, 0, 0, 0, 0, 0};

    const int ldRow = tid >> 1;
    const int ldCol = (tid & 1) * 2;
    const unsigned char* aPtr = Aq + (size_t)(mBlock + ldRow) * K + ldCol * 16;
    const unsigned char* bPtr = Bq + (size_t)(nBlock + ldRow) * K + ldCol * 16;

    {
        uint4 a0 = *(const uint4*)(aPtr);
        uint4 a1 = *(const uint4*)(aPtr + 16);
        uint4 b0 = *(const uint4*)(bPtr);
        uint4 b1 = *(const uint4*)(bPtr + 16);
        As[0][ldRow][ldCol] = a0; As[0][ldRow][ldCol + 1] = a1;
        Bs[0][ldRow][ldCol] = b0; Bs[0][ldRow][ldCol + 1] = b1;
    }
    __syncthreads();

    int buf = 0;
    for (int k = 0; k < K; k += BK) {
        const int kn = k + BK;
        uint4 na0, na1, nb0, nb1;
        const bool more = kn < K;
        if (more) {
            na0 = *(const uint4*)(aPtr + kn);
            na1 = *(const uint4*)(aPtr + kn + 16);
            nb0 = *(const uint4*)(bPtr + kn);
            nb1 = *(const uint4*)(bPtr + kn + 16);
        }

#pragma unroll
        for (int mi = 0; mi < 2; ++mi) {
            const unsigned char* ar =
                (const unsigned char*)&As[buf][waveM + mi * 16 + laneLo][0];
            uint2 q0 = *(const uint2*)(ar + laneHi * 16);
            uint2 q1v = *(const uint2*)(ar + 8 + laneHi * 16);
            uint2 q2 = *(const uint2*)(ar + 32 + laneHi * 16);
            uint2 q3 = *(const uint2*)(ar + 40 + laneHi * 16);
            v8i afrag;
            afrag[0] = (int)q0.x; afrag[1] = (int)q0.y;
            afrag[2] = (int)q1v.x; afrag[3] = (int)q1v.y;
            afrag[4] = (int)q2.x; afrag[5] = (int)q2.y;
            afrag[6] = (int)q3.x; afrag[7] = (int)q3.y;
#pragma unroll
            for (int ni = 0; ni < 4; ++ni) {
                const unsigned char* br =
                    (const unsigned char*)&Bs[buf][waveN + ni * 16 + laneLo][0];
                uint4 p0 = *(const uint4*)(br + laneHi * 16);
                uint4 p1 = *(const uint4*)(br + 32 + laneHi * 16);
                v8i bfrag;
                bfrag[0] = (int)p0.x; bfrag[1] = (int)p0.y;
                bfrag[2] = (int)p0.z; bfrag[3] = (int)p0.w;
                bfrag[4] = (int)p1.x; bfrag[5] = (int)p1.y;
                bfrag[6] = (int)p1.z; bfrag[7] = (int)p1.w;
                acc[mi][ni] = __builtin_amdgcn_wmma_i32_16x16x64_iu8(
                     false, afrag,  false, bfrag,
                    acc[mi][ni],  false,  false);
            }
        }

        if (more) {
            As[buf ^ 1][ldRow][ldCol] = na0; As[buf ^ 1][ldRow][ldCol + 1] = na1;
            Bs[buf ^ 1][ldRow][ldCol] = nb0; Bs[buf ^ 1][ldRow][ldCol + 1] = nb1;
            __syncthreads();
            buf ^= 1;
        }
    }

    const float S1S2 = scal[S_S1S2];
    const float Z1 = scal[S_Z1];
    const float Z2 = scal[S_Z2];
    const float Z1Z2K = scal[S_Z1Z2K];

#pragma unroll
    for (int mi = 0; mi < 2; ++mi) {
        const int mBase = mBlock + waveM + mi * 16 + laneHi * 8;
        float zx[8];
#pragma unroll
        for (int r = 0; r < 8; ++r) zx[r] = Z1 * (float)rowX[mBase + r];
#pragma unroll
        for (int ni = 0; ni < 4; ++ni) {
            const int n = nBlock + waveN + ni * 16 + laneLo;
            const float qb = rintf(bias[n] / S1S2);
            const float zw = Z2 * (float)rowW[n];
            const float cn = qb - zw + Z1Z2K;
#pragma unroll
            for (int r = 0; r < 8; ++r) {
                float v = (float)acc[mi][ni][r] + cn - zx[r];
                out[(size_t)(mBase + r) * N + n] = S1S2 * v;
            }
        }
    }
}

extern "C" void kernel_launch(void* const* d_in, const int* in_sizes, int n_in,
                              void* d_out, int out_size, void* d_ws, size_t ws_size,
                              hipStream_t stream) {
    const float* x = (const float*)d_in[0];
    const float* W = (const float*)d_in[1];
    const float* b = (const float*)d_in[2];
    float* out = (float*)d_out;

    const int N = in_sizes[2];
    const int K = in_sizes[1] / N;
    const int M = in_sizes[0] / K;

    char* ws = (char*)d_ws;
    unsigned* enc = (unsigned*)(ws + OFF_ENC);
    float* scal = (float*)(ws + OFF_SCAL);
    int* rowW = (int*)(ws + OFF_ROWW);
    int* rowX = (int*)(ws + OFF_ROWX);
    unsigned char* q1 = (unsigned char*)(ws + OFF_Q1);
    unsigned char* xq = (unsigned char*)(ws + OFF_XQ);

    init_scalars<<<1, 1, 0, stream>>>(enc);

    long long nW4 = (long long)N * K / 4;
    long long nX4 = (long long)M * K / 4;
    int blkW = (int)((nW4 + 255) / 256); if (blkW > 4096) blkW = 4096;
    int blkX = (int)((nX4 + 255) / 256); if (blkX > 8192) blkX = 8192;
    minmax_kernel<<<blkW, 256, 0, stream>>>(W, nW4, enc + 0, enc + 1);
    minmax_kernel<<<blkX, 256, 0, stream>>>(x, nX4, enc + 2, enc + 3);

    params_kernel<<<1, 1, 0, stream>>>(enc, scal, K);

    quantW_kernel<<<N, 256, 0, stream>>>(W, q1, rowW, scal, K);
    quantX_kernel<<<M, 256, 0, stream>>>(x, xq, rowX, scal, K);

    dim3 grid(N / BN, M / BM);
    gemm_iu8_kernel<<<grid, 256, 0, stream>>>(xq, q1, rowX, rowW, b, scal, out,
                                              M, N, K);
}
